// GNNEmbedder_46677704573058
// MI455X (gfx1250) — compile-verified, matrix operand order checked by reading, not run on hardware
//
#include <hip/hip_runtime.h>
#include <hip/hip_bf16.h>

// GAT-2 + MLP + mean-pool for MI455X (gfx1250, wave32).
// GEMMs use v_wmma_f32_16x16x32_bf16 (f32 accumulate). Edge softmax/scatter
// phases are L2-resident atomics (whole working set < 192MB L2).

typedef __attribute__((ext_vector_type(16))) __bf16 v16bf;
typedef __attribute__((ext_vector_type(8)))  float  v8f;

// ---- monotonic float<->uint map for atomicMax on floats ----
__device__ __forceinline__ unsigned f2ord(float f) {
    unsigned u = __float_as_uint(f);
    return (u & 0x80000000u) ? ~u : (u | 0x80000000u);
}
__device__ __forceinline__ float ord2f(unsigned o) {
    return (o & 0x80000000u) ? __uint_as_float(o ^ 0x80000000u)
                             : __uint_as_float(~o);
}
#define ORD_NEG_INF 0x007FFFFFu   // f2ord(-inf)

// ---------------------------------------------------------------------------
// GEMM: C[M,64] = A[M,64] @ W[64,64] (+bias)(+relu), bf16 WMMA, f32 accum.
// One wave computes a 16x64 strip: 2 A-fragments (kb=0,32) reused over 4
// column tiles; 2 chained v_wmma_f32_16x16x32_bf16 per tile.
// FLAGS (compile-time): bit0 = add bias, bit1 = relu
// ---------------------------------------------------------------------------
template <int FLAGS>
__global__ __launch_bounds__(256) void gemm64_wmma(
    const float* __restrict__ A, const float* __restrict__ W,
    const float* __restrict__ bias, float* __restrict__ C,
    int M)
{
    const int wave  = threadIdx.x >> 5;
    const int lane  = threadIdx.x & 31;
    const int strip = blockIdx.x * 8 + wave;
    const int mbase = strip * 16;
    if (mbase >= M) return;                 // wave-uniform: EXEC stays full
    const bool full = (mbase + 16 <= M);    // wave-uniform fast path

    const int lrow = lane & 15;
    const int koff = (lane < 16) ? 0 : 8;   // ISA 16-bit A/B lane->K mapping

    int m = mbase + lrow;
    if (m >= M) m = M - 1;                  // clamp keeps EXEC all-ones
    const float* arow = A + (size_t)m * 64;

    v16bf a0, a1;
#pragma unroll
    for (int j = 0; j < 8; ++j) {
        a0[j]     = (__bf16)arow[ 0 + koff + j];
        a0[8 + j] = (__bf16)arow[16 + koff + j];
        a1[j]     = (__bf16)arow[32 + koff + j];
        a1[8 + j] = (__bf16)arow[48 + koff + j];
    }

#pragma unroll
    for (int t = 0; t < 4; ++t) {
        const int nc   = t * 16;
        const int ncol = nc + lrow;
        v16bf b0, b1;
#pragma unroll
        for (int j = 0; j < 8; ++j) {
            b0[j]     = (__bf16)W[( 0 + koff + j) * 64 + ncol];
            b0[8 + j] = (__bf16)W[(16 + koff + j) * 64 + ncol];
            b1[j]     = (__bf16)W[(32 + koff + j) * 64 + ncol];
            b1[8 + j] = (__bf16)W[(48 + koff + j) * 64 + ncol];
        }
        v8f c = {};
        c = __builtin_amdgcn_wmma_f32_16x16x32_bf16(false, a0, false, b0,
                                                    (short)0, c, false, false);
        c = __builtin_amdgcn_wmma_f32_16x16x32_bf16(false, a1, false, b1,
                                                    (short)0, c, false, false);

        const int col   = nc + lrow;
        const int rbase = (lane < 16) ? mbase : (mbase + 8);
        const float bv  = (FLAGS & 1) ? bias[col] : 0.0f;
        float* cptr = C + (size_t)rbase * 64 + col;
        if (full) {
#pragma unroll
            for (int r = 0; r < 8; ++r) {
                float v = c[r] + bv;
                if (FLAGS & 2) v = v > 0.0f ? v : 0.0f;
                cptr[(size_t)r * 64] = v;
            }
        } else {
#pragma unroll
            for (int r = 0; r < 8; ++r) {
                if (rbase + r < M) {
                    float v = c[r] + bv;
                    if (FLAGS & 2) v = v > 0.0f ? v : 0.0f;
                    cptr[(size_t)r * 64] = v;
                }
            }
        }
    }
}

// ---------------------------------------------------------------------------
// a_src[n] = xs[n,:]·att_src ; a_dst[n] = xs[n,:]·att_dst. One wave per node.
// ---------------------------------------------------------------------------
__global__ __launch_bounds__(256) void attn_dots(
    const float* __restrict__ xs, const float* __restrict__ att_s,
    const float* __restrict__ att_d, float* __restrict__ a_s,
    float* __restrict__ a_d, int N)
{
    const int wave = threadIdx.x >> 5;
    const int lane = threadIdx.x & 31;
    const int n = blockIdx.x * 8 + wave;
    if (n >= N) return;
    const float2 v  = *(const float2*)(xs + (size_t)n * 64 + lane * 2);
    const float2 as = *(const float2*)(att_s + lane * 2);
    const float2 ad = *(const float2*)(att_d + lane * 2);
    float ss = v.x * as.x + v.y * as.y;
    float sd = v.x * ad.x + v.y * ad.y;
#pragma unroll
    for (int off = 16; off; off >>= 1) {
        ss += __shfl_xor(ss, off, 32);
        sd += __shfl_xor(sd, off, 32);
    }
    if (lane == 0) { a_s[n] = ss; a_d[n] = sd; }
}

__device__ __forceinline__ void edge_sd(const int* __restrict__ ei, int E,
                                        int i, int& s, int& d) {
    if (i < E) { s = ei[i]; d = ei[E + i]; }
    else       { s = i - E; d = i - E; }       // self-loops appended
}

// pass1: e = leaky_relu(a_src[s] + a_dst[d]); segment-max into mmax[d]
__global__ void edge_pass1(const int* __restrict__ ei, int E, int EA,
                           const float* __restrict__ a_s,
                           const float* __restrict__ a_d,
                           float* __restrict__ e, unsigned* __restrict__ mmax)
{
    const int i = blockIdx.x * blockDim.x + threadIdx.x;
    if (i >= EA) return;
    int s, d; edge_sd(ei, E, i, s, d);
    float v = a_s[s] + a_d[d];
    v = (v > 0.0f) ? v : 0.2f * v;
    e[i] = v;
    atomicMax(&mmax[d], f2ord(v));
}

// pass2: p = exp(e - m[d]) (in place); segment-sum into denom[d]
__global__ void edge_pass2(const int* __restrict__ ei, int E, int EA,
                           const unsigned* __restrict__ mmax,
                           float* __restrict__ e, float* __restrict__ denom)
{
    const int i = blockIdx.x * blockDim.x + threadIdx.x;
    if (i >= EA) return;
    int s, d; edge_sd(ei, E, i, s, d);
    const float p = __expf(e[i] - ord2f(mmax[d]));
    e[i] = p;
    atomicAdd(&denom[d], p);
}

// pass3: acc[d,:] += (p/denom[d]) * xs[s,:]. One wave per edge, lane = 2 feats.
__global__ __launch_bounds__(256) void edge_pass3(
    const int* __restrict__ ei, int E, int EA,
    const float* __restrict__ p, const float* __restrict__ denom,
    const float* __restrict__ xs, float* __restrict__ acc)
{
    const int wave = threadIdx.x >> 5;
    const int lane = threadIdx.x & 31;
    const int i = blockIdx.x * 8 + wave;
    if (i >= EA) return;
    int s, d; edge_sd(ei, E, i, s, d);
    const float alpha = p[i] / denom[d];
    const float2 v = *(const float2*)(xs + (size_t)s * 64 + lane * 2);
    float* o = acc + (size_t)d * 64 + lane * 2;
    atomicAdd(o,     alpha * v.x);
    atomicAdd(o + 1, alpha * v.y);
}

// ---- small utility kernels ----
__global__ void fill_f32(float* p, float v, int n) {
    int i = blockIdx.x * blockDim.x + threadIdx.x;
    if (i < n) p[i] = v;
}
__global__ void fill_u32(unsigned* p, unsigned v, int n) {
    int i = blockIdx.x * blockDim.x + threadIdx.x;
    if (i < n) p[i] = v;
}
__global__ void bias_init(float* acc, const float* __restrict__ b, int n64) {
    int i = blockIdx.x * blockDim.x + threadIdx.x;
    if (i < n64) acc[i] = b[i & 63];
}
__global__ void relu_ip(float* p, int n) {
    int i = blockIdx.x * blockDim.x + threadIdx.x;
    if (i < n) { float v = p[i]; p[i] = v > 0.0f ? v : 0.0f; }
}

// pooling
__global__ void pool_acc(const float* __restrict__ nodes,
                         const int* __restrict__ batch,
                         float* __restrict__ pool, float* __restrict__ cnt,
                         int N)
{
    int i = blockIdx.x * blockDim.x + threadIdx.x;
    if (i >= N * 64) return;
    const int n = i >> 6, f = i & 63;
    const int g = batch[n];
    atomicAdd(&pool[g * 64 + f], nodes[i]);
    if (f == 0) atomicAdd(&cnt[g], 1.0f);
}
__global__ void pool_div(const float* __restrict__ pool,
                         const float* __restrict__ cnt,
                         float* __restrict__ out, int n)
{
    int i = blockIdx.x * blockDim.x + threadIdx.x;
    if (i < n) out[i] = pool[i] / fmaxf(cnt[i >> 6], 1.0f);
}

// ---------------------------------------------------------------------------
extern "C" void kernel_launch(void* const* d_in, const int* in_sizes, int n_in,
                              void* d_out, int out_size, void* d_ws, size_t ws_size,
                              hipStream_t stream)
{
    const float* x        = (const float*)d_in[0];
    const int*   ei       = (const int*)  d_in[1];
    const int*   batch    = (const int*)  d_in[2];
    const float* W1       = (const float*)d_in[3];
    const float* att_s1   = (const float*)d_in[4];
    const float* att_d1   = (const float*)d_in[5];
    const float* b1       = (const float*)d_in[6];
    const float* W2       = (const float*)d_in[7];
    const float* att_s2   = (const float*)d_in[8];
    const float* att_d2   = (const float*)d_in[9];
    const float* b2       = (const float*)d_in[10];
    const float* mW1      = (const float*)d_in[11];
    const float* mb1      = (const float*)d_in[12];
    const float* mW2      = (const float*)d_in[13];
    const float* mb2      = (const float*)d_in[14];

    const int N  = in_sizes[2];
    const int E  = in_sizes[1] / 2;
    const int EA = E + N;              // edges + self-loops
    const int G  = out_size / 64;

    // workspace layout (floats)
    const size_t f64N = (size_t)N * 64;
    float*    bufA  = (float*)d_ws;            // xs / mlp tmp  [N*64]
    float*    bufB  = bufA + f64N;             // acc / h       [N*64]
    float*    asrc  = bufB + f64N;             // [N]
    float*    adst  = asrc + N;                // [N]
    unsigned* mmax  = (unsigned*)(adst + N);   // [N]
    float*    denom = (float*)(mmax + N);      // [N]
    float*    ep    = denom + N;               // [EA] e then p
    float*    pool  = ep + EA;                 // [G*64]
    float*    cnt   = pool + (size_t)G * 64;   // [G]

    auto cdiv = [](long a, long b) { return (int)((a + b - 1) / b); };
    const int strips = cdiv(N, 16);
    const int gemmBlk = cdiv(strips, 8);
    const int n64 = N * 64;

    // ---- GAT layer 1 ----
    gemm64_wmma<0><<<gemmBlk, 256, 0, stream>>>(x, W1, nullptr, bufA, N);
    attn_dots  <<<cdiv(N, 8), 256, 0, stream>>>(bufA, att_s1, att_d1, asrc, adst, N);
    fill_u32   <<<cdiv(N, 256), 256, 0, stream>>>(mmax, ORD_NEG_INF, N);
    fill_f32   <<<cdiv(N, 256), 256, 0, stream>>>(denom, 0.0f, N);
    bias_init  <<<cdiv(n64, 256), 256, 0, stream>>>(bufB, b1, n64);
    edge_pass1 <<<cdiv(EA, 256), 256, 0, stream>>>(ei, E, EA, asrc, adst, ep, mmax);
    edge_pass2 <<<cdiv(EA, 256), 256, 0, stream>>>(ei, E, EA, mmax, ep, denom);
    edge_pass3 <<<cdiv(EA, 8), 256, 0, stream>>>(ei, E, EA, ep, denom, bufA, bufB);
    relu_ip    <<<cdiv(n64, 256), 256, 0, stream>>>(bufB, n64);   // h1 in bufB

    // ---- GAT layer 2 ----
    gemm64_wmma<0><<<gemmBlk, 256, 0, stream>>>(bufB, W2, nullptr, bufA, N);
    attn_dots  <<<cdiv(N, 8), 256, 0, stream>>>(bufA, att_s2, att_d2, asrc, adst, N);
    fill_u32   <<<cdiv(N, 256), 256, 0, stream>>>(mmax, ORD_NEG_INF, N);
    fill_f32   <<<cdiv(N, 256), 256, 0, stream>>>(denom, 0.0f, N);
    bias_init  <<<cdiv(n64, 256), 256, 0, stream>>>(bufB, b2, n64);
    edge_pass1 <<<cdiv(EA, 256), 256, 0, stream>>>(ei, E, EA, asrc, adst, ep, mmax);
    edge_pass2 <<<cdiv(EA, 256), 256, 0, stream>>>(ei, E, EA, mmax, ep, denom);
    edge_pass3 <<<cdiv(EA, 8), 256, 0, stream>>>(ei, E, EA, ep, denom, bufA, bufB);
    relu_ip    <<<cdiv(n64, 256), 256, 0, stream>>>(bufB, n64);   // h2 in bufB

    // ---- MLP ----
    gemm64_wmma<3><<<gemmBlk, 256, 0, stream>>>(bufB, mW1, mb1, bufA, N); // bias+relu
    gemm64_wmma<1><<<gemmBlk, 256, 0, stream>>>(bufA, mW2, mb2, bufB, N); // bias

    // ---- global mean pool ----
    fill_f32<<<cdiv(G * 64, 256), 256, 0, stream>>>(pool, 0.0f, G * 64);
    fill_f32<<<cdiv(G, 256), 256, 0, stream>>>(cnt, 0.0f, G);
    pool_acc<<<cdiv(n64, 256), 256, 0, stream>>>(bufB, batch, pool, cnt, N);
    pool_div<<<cdiv(G * 64, 256), 256, 0, stream>>>(pool, cnt, (float*)d_out, G * 64);
}
